// LSTMModel_5317169512601
// MI455X (gfx1250) — matrix operand order repaired and checked by reading, not compiled, not run on hardware
//
#include <hip/hip_runtime.h>
#include <hip/hip_bf16.h>

typedef __attribute__((ext_vector_type(16))) _Float16 v16h;
typedef __attribute__((ext_vector_type(8)))  _Float16 v8h;
typedef __attribute__((ext_vector_type(8)))  float    v8f;

// Problem constants (match reference)
constexpr int Tt  = 512;   // timesteps
constexpr int Ii  = 3;     // input features
constexpr int Hh  = 50;    // hidden
constexpr int G4  = 200;   // 4*H
// Tiling
constexpr int TB  = 64;    // batch rows per block
constexpr int Hp  = 64;    // padded K (50 -> 64)
constexpr int Np  = 208;   // padded N (200 -> 208)
constexpr int NT  = 13;    // N tiles of 16
constexpr int GS  = 210;   // gates LDS row stride (floats), avoids bank aliasing
constexpr int NTH = 128;   // threads per block (4 waves)
constexpr int QP  = (TB * Hh) / NTH; // 25 (row,h) pairs per thread

__device__ __forceinline__ float sigm(float v) { return 1.0f / (1.0f + __expf(-v)); }

// Load a 16-half WMMA fragment as two 16-byte LDS loads.
__device__ __forceinline__ v16h ldfrag(const _Float16* p, int second_off) {
  v8h a = *(const v8h*)p;
  v8h b = *(const v8h*)(p + second_off);
  return __builtin_shufflevector(a, b, 0,1,2,3,4,5,6,7,8,9,10,11,12,13,14,15);
}

__device__ __forceinline__ v8f wmma_f16(v16h a, v16h b, v8f c) {
  return __builtin_amdgcn_wmma_f32_16x16x32_f16(false, a, false, b, (short)0, c, false, false);
}

__global__ __launch_bounds__(NTH, 1) void lstm2_fused(
    const float* __restrict__ x,
    const float* __restrict__ Wih0, const float* __restrict__ Whh0,
    const float* __restrict__ bih0, const float* __restrict__ bhh0,
    const float* __restrict__ Wih1, const float* __restrict__ Whh1,
    const float* __restrict__ bih1, const float* __restrict__ bhh1,
    const float* __restrict__ Wfc,  const float* __restrict__ bfc,
    float* __restrict__ out) {

  // LDS: weights (f16, N-major [Np][Hp]), h state (f16 [TB][Hp]), gates (f32)
  __shared__ __attribute__((aligned(16))) _Float16 sWhh0[Np * Hp];
  __shared__ __attribute__((aligned(16))) _Float16 sWih1[Np * Hp];
  __shared__ __attribute__((aligned(16))) _Float16 sWhh1[Np * Hp];
  __shared__ __attribute__((aligned(16))) _Float16 sH0[TB * Hp];
  __shared__ __attribute__((aligned(16))) _Float16 sH1[TB * Hp];
  __shared__ __attribute__((aligned(16))) float    sGates[TB * GS];
  __shared__ float sX[TB * 4];
  __shared__ float sWi0[G4 * 3];
  __shared__ float sB0[G4];
  __shared__ float sB1[G4];

  const int tid   = threadIdx.x;
  const int rbase = blockIdx.x * TB;

  // ---- one-time staging ----
  for (int idx = tid; idx < Np * Hp; idx += NTH) {
    const int col = idx >> 6;        // padded N index
    const int k   = idx & 63;        // padded K index
    const bool ok = (col < G4) && (k < Hh);
    const _Float16 z = (_Float16)0.0f;
    sWhh0[idx] = ok ? (_Float16)Whh0[col * Hh + k] : z;
    sWih1[idx] = ok ? (_Float16)Wih1[col * Hh + k] : z;
    sWhh1[idx] = ok ? (_Float16)Whh1[col * Hh + k] : z;
  }
  for (int idx = tid; idx < TB * Hp; idx += NTH) {
    sH0[idx] = (_Float16)0.0f;
    sH1[idx] = (_Float16)0.0f;
  }
  for (int idx = tid; idx < G4; idx += NTH) {
    sB0[idx] = bih0[idx] + bhh0[idx];
    sB1[idx] = bih1[idx] + bhh1[idx];
  }
  for (int idx = tid; idx < G4 * 3; idx += NTH) sWi0[idx] = Wih0[idx];
  __syncthreads();

  // per-thread cell state and (row,h) map
  float c0[QP], c1[QP];
  int   rw[QP], hh[QP];
#pragma unroll
  for (int q = 0; q < QP; ++q) {
    c0[q] = 0.0f; c1[q] = 0.0f;
    const int idx = q * NTH + tid;   // 0..3199
    rw[q] = idx / Hh;
    hh[q] = idx - rw[q] * Hh;
  }

  // wave geometry: wave w owns M-tile w (rows w*16 .. w*16+15)
  const int w    = tid >> 5;
  const int l    = tid & 31;
  const int lo16 = l & 15;
  const int hi   = (l >> 4) & 1;
  const int Mrow = w * 16 + lo16;

  const _Float16* aBase0 = sH0 + Mrow * Hp;
  const _Float16* aBase1 = sH1 + Mrow * Hp;

#pragma unroll 1
  for (int t = 0; t < Tt; ++t) {
    // ===== Phase A: stage x[:, t, :] and layer-0 recurrent GEMM =====
    if (tid < TB * Ii) {
      const int r = tid / 3, d = tid - r * 3;
      sX[r * 4 + d] = x[((size_t)(rbase + r) * Tt + t) * Ii + d];
    }
    {
      v16h a0 = ldfrag(aBase0 + 0 * 32 + hi * 8, 16);
      v16h a1 = ldfrag(aBase0 + 1 * 32 + hi * 8, 16);
#pragma unroll
      for (int n = 0; n < NT; ++n) {
        const _Float16* bp = sWhh0 + (n * 16 + lo16) * Hp;
        v16h b0 = ldfrag(bp + 0 * 32 + hi * 8, 16);
        v16h b1 = ldfrag(bp + 1 * 32 + hi * 8, 16);
        v8f acc = {};
        acc = wmma_f16(a0, b0, acc);
        acc = wmma_f16(a1, b1, acc);
        float* gp = sGates + (w * 16 + hi * 8) * GS + n * 16 + lo16;
#pragma unroll
        for (int j = 0; j < 8; ++j) gp[j * GS] = acc[j];
      }
    }
    __syncthreads();

    // ===== Phase B: layer-0 elementwise (adds x-part + bias) =====
#pragma unroll
    for (int q = 0; q < QP; ++q) {
      const int r = rw[q], h = hh[q];
      const float x0 = sX[r * 4 + 0], x1 = sX[r * 4 + 1], x2 = sX[r * 4 + 2];
      const float* g = sGates + r * GS;
      const int ci = h, cf = h + Hh, cg = h + 2 * Hh, co = h + 3 * Hh;
      float gi = g[ci] + sB0[ci] + x0 * sWi0[ci * 3] + x1 * sWi0[ci * 3 + 1] + x2 * sWi0[ci * 3 + 2];
      float gf = g[cf] + sB0[cf] + x0 * sWi0[cf * 3] + x1 * sWi0[cf * 3 + 1] + x2 * sWi0[cf * 3 + 2];
      float gg = g[cg] + sB0[cg] + x0 * sWi0[cg * 3] + x1 * sWi0[cg * 3 + 1] + x2 * sWi0[cg * 3 + 2];
      float go = g[co] + sB0[co] + x0 * sWi0[co * 3] + x1 * sWi0[co * 3 + 1] + x2 * sWi0[co * 3 + 2];
      const float iv = sigm(gi), fv = sigm(gf), ov = sigm(go), gv = tanhf(gg);
      c0[q] = fv * c0[q] + iv * gv;
      sH0[r * Hp + h] = (_Float16)(ov * tanhf(c0[q]));
    }
    __syncthreads();

    // ===== Phase C: layer-1 GEMMs (input from h0_new, recurrent from h1) =====
    {
      v16h a00 = ldfrag(aBase0 + 0 * 32 + hi * 8, 16);
      v16h a01 = ldfrag(aBase0 + 1 * 32 + hi * 8, 16);
      v16h a10 = ldfrag(aBase1 + 0 * 32 + hi * 8, 16);
      v16h a11 = ldfrag(aBase1 + 1 * 32 + hi * 8, 16);
#pragma unroll
      for (int n = 0; n < NT; ++n) {
        const _Float16* bpi = sWih1 + (n * 16 + lo16) * Hp;
        const _Float16* bph = sWhh1 + (n * 16 + lo16) * Hp;
        v16h bi0 = ldfrag(bpi + 0 * 32 + hi * 8, 16);
        v16h bi1 = ldfrag(bpi + 1 * 32 + hi * 8, 16);
        v16h bh0 = ldfrag(bph + 0 * 32 + hi * 8, 16);
        v16h bh1 = ldfrag(bph + 1 * 32 + hi * 8, 16);
        v8f acc = {};
        acc = wmma_f16(a00, bi0, acc);
        acc = wmma_f16(a01, bi1, acc);
        acc = wmma_f16(a10, bh0, acc);
        acc = wmma_f16(a11, bh1, acc);
        float* gp = sGates + (w * 16 + hi * 8) * GS + n * 16 + lo16;
#pragma unroll
        for (int j = 0; j < 8; ++j) gp[j * GS] = acc[j];
      }
    }
    __syncthreads();

    // ===== Phase D: layer-1 elementwise =====
#pragma unroll
    for (int q = 0; q < QP; ++q) {
      const int r = rw[q], h = hh[q];
      const float* g = sGates + r * GS;
      float gi = g[h] + sB1[h];
      float gf = g[h + Hh] + sB1[h + Hh];
      float gg = g[h + 2 * Hh] + sB1[h + 2 * Hh];
      float go = g[h + 3 * Hh] + sB1[h + 3 * Hh];
      const float iv = sigm(gi), fv = sigm(gf), ov = sigm(go), gv = tanhf(gg);
      c1[q] = fv * c1[q] + iv * gv;
      sH1[r * Hp + h] = (_Float16)(ov * tanhf(c1[q]));
    }
    __syncthreads();
  }

  // ===== FC epilogue: out[row] = h1_last . W_fc + b_fc =====
  if (tid < TB) {
    float acc = bfc[0];
#pragma unroll 1
    for (int h = 0; h < Hh; ++h) acc += (float)sH1[tid * Hp + h] * Wfc[h];
    out[rbase + tid] = acc;
  }
}

extern "C" void kernel_launch(void* const* d_in, const int* in_sizes, int n_in,
                              void* d_out, int out_size, void* d_ws, size_t ws_size,
                              hipStream_t stream) {
  const float* x    = (const float*)d_in[0];
  const float* Wih0 = (const float*)d_in[1];
  const float* Whh0 = (const float*)d_in[2];
  const float* bih0 = (const float*)d_in[3];
  const float* bhh0 = (const float*)d_in[4];
  const float* Wih1 = (const float*)d_in[5];
  const float* Whh1 = (const float*)d_in[6];
  const float* bih1 = (const float*)d_in[7];
  const float* bhh1 = (const float*)d_in[8];
  const float* Wfc  = (const float*)d_in[9];
  const float* bfc  = (const float*)d_in[10];
  float* out = (float*)d_out;

  const int Bsz = 4096;
  dim3 grid(Bsz / TB), block(NTH);
  lstm2_fused<<<grid, block, 0, stream>>>(x, Wih0, Whh0, bih0, bhh0,
                                          Wih1, Whh1, bih1, bhh1, Wfc, bfc, out);
  (void)in_sizes; (void)n_in; (void)out_size; (void)d_ws; (void)ws_size;
}
